// MambaMoELayer_35734127903108
// MI455X (gfx1250) — compile-verified, matrix operand order checked by reading, not run on hardware
//
#include <hip/hip_runtime.h>
#include <hip/hip_bf16.h>
#include <math.h>

#define DIM_   1024
#define SEQ_   2048
#define DINNER 4096
#define DSTATE 16
#define DTRANK 64
#define XPOUT  96      // DTRANK + 2*DSTATE
#define NEXP   8
#define HID_   4096
#define DCONV  4

typedef __bf16 bf16_t;
typedef bf16_t v16bf __attribute__((ext_vector_type(16)));
typedef float  v8f   __attribute__((ext_vector_type(8)));
typedef float  v4f   __attribute__((ext_vector_type(4)));

__device__ __forceinline__ bf16_t f2bf(float f) {
  unsigned u = __builtin_bit_cast(unsigned, f);
  unsigned r = u + 0x7fffu + ((u >> 16) & 1u);   // round-to-nearest-even
  unsigned short s = (unsigned short)(r >> 16);
  return __builtin_bit_cast(bf16_t, s);
}

__device__ __forceinline__ float block_sum_256(float v) {
  #pragma unroll
  for (int off = 16; off > 0; off >>= 1) v += __shfl_down(v, off, 32);
  __shared__ float red[8];
  __shared__ float tot;
  int lane = threadIdx.x & 31, w = threadIdx.x >> 5;
  if (lane == 0) red[w] = v;
  __syncthreads();
  if (threadIdx.x == 0) {
    float s = 0.f;
    #pragma unroll
    for (int i = 0; i < 8; ++i) s += red[i];
    tot = s;
  }
  __syncthreads();
  return tot;
}

// f32 -> bf16 bulk conversion (weights staged once per launch)
__global__ __launch_bounds__(256)
void cvt_bf16_kernel(const float* __restrict__ in, bf16_t* __restrict__ o, long n) {
  long i = ((long)blockIdx.x * 256 + threadIdx.x) * 4;
  if (i + 4 <= n) {
    v4f v = *(const v4f*)(in + i);
    o[i]     = f2bf(v[0]);
    o[i + 1] = f2bf(v[1]);
    o[i + 2] = f2bf(v[2]);
    o[i + 3] = f2bf(v[3]);
  }
}

// ---------------------------------------------------------------------------
// Wave-tile bf16 WMMA GEMM:  C[M,N] (+)= coef * (A[M,K] * W[N,K]^T)
// A, W both bf16 row-major (K contiguous).
// B tile (TILE_N x 32k, shared by the block's 4 M-stacked waves) is staged
// into LDS with double-buffered global_load_async_to_lds_b128 (ASYNCcnt),
// handed off with s_wait_asynccnt + barrier; fragments read via ds loads.
// Each wave: 16 rows x NPW*16 cols, fp32 accum via v_wmma_f32_16x16x32_bf16.
// ---------------------------------------------------------------------------
template <int NPW>
__global__ __launch_bounds__(128)
void gemm_bf16_wmma(const bf16_t* __restrict__ A, const bf16_t* __restrict__ Wb,
                    float* __restrict__ C, int M, int N, int K,
                    const int* __restrict__ gatherA,
                    const int* __restrict__ scatterC,
                    const int* __restrict__ mcount,
                    const float* __restrict__ coef, int coefStride,
                    int accumulate) {
  constexpr int TILE_N    = NPW * 16;
  constexpr int BUF_BYTES = TILE_N * 64;            // 32 k-elems * 2B per row
  constexpr int NLOADS    = BUF_BYTES / (128 * 16); // b128 async loads / thread
  __shared__ __align__(16) bf16_t ldsB[2][TILE_N * 32];

  const int lane = threadIdx.x & 31;
  const int wave = threadIdx.x >> 5;
  const int Meff = mcount ? ((*mcount < M) ? *mcount : M) : M;
  if ((int)blockIdx.x * 64 >= Meff) return;   // uniform exit, before barriers
  const int n0 = blockIdx.y * TILE_N;
  const int m0 = blockIdx.x * 64 + wave * 16;

  const int  mrow   = m0 + (lane & 15);
  const bool mvalid = mrow < Meff;
  const int  srcRow = mvalid ? mrow : (Meff - 1);   // Meff >= 1 here
  const int  arow   = gatherA ? gatherA[srcRow] : srcRow;
  const bf16_t* Ap = A + (size_t)arow * K + ((lane >> 4) << 4);

  const unsigned ldsBase = (unsigned)(uintptr_t)(&ldsB[0][0]);

  auto issueB = [&](int buf, int k0) {
    #pragma unroll
    for (int it = 0; it < NLOADS; ++it) {
      int bidx = ((int)threadIdx.x + it * 128) * 16;  // byte index in tile
      int row  = bidx >> 6;                           // 64B per row
      int col  = bidx & 63;
      int nr   = n0 + row; if (nr > N - 1) nr = N - 1;
      const bf16_t* g = Wb + (size_t)nr * K + k0 + (col >> 1);
      unsigned laddr = ldsBase + (unsigned)(buf * BUF_BYTES + bidx);
      asm volatile("global_load_async_to_lds_b128 %0, %1, off"
                   :: "v"(laddr), "v"((unsigned long long)(uintptr_t)g)
                   : "memory");
    }
  };

  v8f acc[NPW];
  #pragma unroll
  for (int j = 0; j < NPW; ++j)
    #pragma unroll
    for (int r = 0; r < 8; ++r) acc[j][r] = 0.f;

  issueB(0, 0);
  const int nk = K / 32;
  for (int ks = 0; ks < nk; ++ks) {
    const int k0  = ks * 32;
    const int buf = ks & 1;
    if (ks + 1 < nk) {
      issueB(buf ^ 1, k0 + 32);
      __builtin_prefetch(Ap + k0 + 32, 0, 1);
      asm volatile("s_wait_asynccnt %0" :: "i"(NLOADS) : "memory");
    } else {
      asm volatile("s_wait_asynccnt 0" ::: "memory");
    }
    __syncthreads();                                  // publish LDS tile
    v16bf a = *(const v16bf*)(Ap + k0);
    #pragma unroll
    for (int j = 0; j < NPW; ++j) {
      const v16bf* bp = (const v16bf*)
          &ldsB[buf][((j * 16 + (lane & 15)) << 5) + ((lane >> 4) << 4)];
      v16bf b = *bp;
      acc[j] = __builtin_amdgcn_wmma_f32_16x16x32_bf16(
          false, a, false, b, (short)0, acc[j], false, false);
    }
    __syncthreads();                                  // done reading buf
  }

  // C/D layout: VGPR r -> M = r + 8*(lane>=16), N = lane & 15
  const int rowBase = m0 + ((lane >> 4) << 3);
  const int colLane = lane & 15;
  #pragma unroll
  for (int j = 0; j < NPW; ++j) {
    int ccol = n0 + j * 16 + colLane;
    if (ccol >= N) continue;
    #pragma unroll
    for (int r = 0; r < 8; ++r) {
      int crow = rowBase + r;
      if (crow >= Meff) continue;
      int orow = scatterC ? scatterC[crow] : crow;
      float v = acc[j][r];
      if (coef) v *= coef[(size_t)orow * coefStride];
      size_t oi = (size_t)orow * N + ccol;
      if (accumulate) C[oi] += v; else C[oi] = v;
    }
  }
}

// ---------------------------------------------------------------------------
// RMSNorm( a (+ b) ) -> f32 and bf16 copies.  One block per token.
// ---------------------------------------------------------------------------
__global__ __launch_bounds__(256)
void addnorm_kernel(const float* __restrict__ a, const float* __restrict__ b,
                    float* __restrict__ of, bf16_t* __restrict__ ob) {
  const int t = blockIdx.x;
  const float* ar = a + (size_t)t * DIM_;
  const float* br = b ? b + (size_t)t * DIM_ : nullptr;
  float ss = 0.f;
  for (int i = threadIdx.x; i < DIM_; i += 256) {
    float v = ar[i] + (br ? br[i] : 0.f);
    ss += v * v;
  }
  float tot = block_sum_256(ss);
  float scale = 32.0f / fmaxf(sqrtf(tot), 1e-12f);   // sqrt(1024)=32
  for (int i = threadIdx.x; i < DIM_; i += 256) {
    float v = (ar[i] + (br ? br[i] : 0.f)) * scale;
    of[(size_t)t * DIM_ + i] = v;
    ob[(size_t)t * DIM_ + i] = f2bf(v);
  }
}

// causal depthwise conv1d (k=4) + SiLU; reads xm half of xr
__global__ __launch_bounds__(256)
void conv_silu_kernel(const float* __restrict__ xr, const float* __restrict__ cw,
                      const float* __restrict__ cb, float* __restrict__ xc,
                      bf16_t* __restrict__ xcb) {
  int idx = blockIdx.x * 256 + threadIdx.x;
  if (idx >= SEQ_ * DINNER) return;
  int t = idx / DINNER, d = idx - t * DINNER;
  float s = cb[d];
  #pragma unroll
  for (int j = 0; j < DCONV; ++j) {
    int tt = t - (DCONV - 1) + j;
    if (tt >= 0) s += xr[(size_t)tt * (2 * DINNER) + d] * cw[d * DCONV + j];
  }
  float v = s / (1.f + __expf(-s));
  xc[idx]  = v;
  xcb[idx] = f2bf(v);
}

__global__ void cvt_dt_kernel(const float* __restrict__ xdbl,
                              bf16_t* __restrict__ dtb) {
  int idx = blockIdx.x * 256 + threadIdx.x;
  if (idx >= SEQ_ * DTRANK) return;
  int t = idx / DTRANK, c = idx - t * DTRANK;
  dtb[idx] = f2bf(xdbl[(size_t)t * XPOUT + c]);
}

__global__ void softplus_kernel(float* __restrict__ delta,
                                const float* __restrict__ bias) {
  int idx = blockIdx.x * 256 + threadIdx.x;
  if (idx >= SEQ_ * DINNER) return;
  int d = idx - (idx / DINNER) * DINNER;
  float xv = delta[idx] + bias[d];
  delta[idx] = fmaxf(xv, 0.f) + log1pf(__expf(-fabsf(xv)));
}

// selective scan: one channel per lane, 16-state in registers,
// B/C broadcast through LDS each timestep
__global__ __launch_bounds__(256)
void scan_kernel(const float* __restrict__ u, const float* __restrict__ delta,
                 const float* __restrict__ xdbl, const float* __restrict__ A_log,
                 const float* __restrict__ Dv, float* __restrict__ y) {
  int d = blockIdx.x * 256 + threadIdx.x;
  float Ac[DSTATE];
  #pragma unroll
  for (int n = 0; n < DSTATE; ++n) Ac[n] = -__expf(A_log[(size_t)d * DSTATE + n]);
  float h[DSTATE];
  #pragma unroll
  for (int n = 0; n < DSTATE; ++n) h[n] = 0.f;
  float Dd = Dv[d];
  __shared__ float sBC[2 * DSTATE];
  for (int t = 0; t < SEQ_; ++t) {
    if (threadIdx.x < 2 * DSTATE)
      sBC[threadIdx.x] = xdbl[(size_t)t * XPOUT + DTRANK + threadIdx.x];
    __syncthreads();
    float dt = delta[(size_t)t * DINNER + d];
    float ut = u[(size_t)t * DINNER + d];
    float accv = 0.f;
    #pragma unroll
    for (int n = 0; n < DSTATE; ++n) {
      float dA = __expf(dt * Ac[n]);
      h[n] = dA * h[n] + dt * sBC[n] * ut;
      accv += h[n] * sBC[DSTATE + n];
    }
    y[(size_t)t * DINNER + d] = accv + ut * Dd;
    __syncthreads();
  }
}

__global__ void gatemul_kernel(const float* __restrict__ ysc,
                               const float* __restrict__ xr,
                               bf16_t* __restrict__ yb) {
  int idx = blockIdx.x * 256 + threadIdx.x;
  if (idx >= SEQ_ * DINNER) return;
  int t = idx / DINNER, d = idx - t * DINNER;
  float r = xr[(size_t)t * (2 * DINNER) + DINNER + d];
  float sil = r / (1.f + __expf(-r));
  yb[idx] = f2bf(ysc[idx] * sil);
}

__global__ void zero_counts_kernel(int* __restrict__ c) {
  if (threadIdx.x < NEXP) c[threadIdx.x] = 0;
}

__global__ void copy_kernel(const float* __restrict__ a, float* __restrict__ o) {
  int idx = blockIdx.x * 256 + threadIdx.x;
  if (idx < SEQ_ * DIM_) o[idx] = a[idx];
}

// MoE router: 8 experts x 32-lane dot products, softmax, top-2, renorm,
// dense coef table + per-expert token lists via atomics
__global__ __launch_bounds__(256)
void router_kernel(const float* __restrict__ xmn, const float* __restrict__ gw,
                   float* __restrict__ coef, int* __restrict__ counts,
                   int* __restrict__ lists) {
  int t = blockIdx.x;
  int lane = threadIdx.x & 31;
  int e = threadIdx.x >> 5;
  float p = 0.f;
  for (int i = lane; i < DIM_; i += 32)
    p += xmn[(size_t)t * DIM_ + i] * gw[e * DIM_ + i];
  #pragma unroll
  for (int off = 16; off > 0; off >>= 1) p += __shfl_down(p, off, 32);
  __shared__ float logit[NEXP];
  if (lane == 0) logit[e] = p;
  __syncthreads();
  if (threadIdx.x == 0) {
    float mx = logit[0];
    for (int i = 1; i < NEXP; ++i) mx = fmaxf(mx, logit[i]);
    float g[NEXP]; float s = 0.f;
    for (int i = 0; i < NEXP; ++i) { g[i] = __expf(logit[i] - mx); s += g[i]; }
    float inv = 1.f / s;
    for (int i = 0; i < NEXP; ++i) g[i] *= inv;
    int i0 = 0;
    for (int i = 1; i < NEXP; ++i) if (g[i] > g[i0]) i0 = i;
    int i1 = (i0 == 0) ? 1 : 0;
    for (int i = 0; i < NEXP; ++i) if (i != i0 && g[i] > g[i1]) i1 = i;
    float r = 1.f / (g[i0] + g[i1]);
    for (int i = 0; i < NEXP; ++i) coef[(size_t)t * NEXP + i] = 0.f;
    coef[(size_t)t * NEXP + i0] = g[i0] * r;
    coef[(size_t)t * NEXP + i1] = g[i1] * r;
    int p0 = atomicAdd(&counts[i0], 1); lists[i0 * SEQ_ + p0] = t;
    int p1 = atomicAdd(&counts[i1], 1); lists[i1 * SEQ_ + p1] = t;
  }
}

__global__ void gelu_kernel(const float* __restrict__ h, bf16_t* __restrict__ hb,
                            const int* __restrict__ count) {
  int idx = blockIdx.x * 256 + threadIdx.x;
  if (idx >= SEQ_ * HID_) return;
  int row = idx / HID_;
  if (row >= *count) return;
  float v = h[idx];
  float g = 0.5f * v * (1.f + erff(v * 0.70710678118f));
  hb[idx] = f2bf(g);
}

extern "C" void kernel_launch(void* const* d_in, const int* in_sizes, int n_in,
                              void* d_out, int out_size, void* d_ws, size_t ws_size,
                              hipStream_t stream) {
  (void)in_sizes; (void)n_in; (void)out_size; (void)ws_size;
  const float* x        = (const float*)d_in[0];
  const float* in_w     = (const float*)d_in[1];   // (8192,1024)
  const float* conv_w   = (const float*)d_in[2];   // (4096,1,4)
  const float* conv_b   = (const float*)d_in[3];   // (4096)
  const float* xproj_w  = (const float*)d_in[4];   // (96,4096)
  const float* dtproj_w = (const float*)d_in[5];   // (4096,64)
  const float* dtproj_b = (const float*)d_in[6];   // (4096)
  const float* A_log    = (const float*)d_in[7];   // (4096,16)
  const float* Dvec     = (const float*)d_in[8];   // (4096)
  const float* out_w    = (const float*)d_in[9];   // (1024,4096)
  const float* gate_w   = (const float*)d_in[10];  // (8,1024)
  const float* w1       = (const float*)d_in[11];  // (8,4096,1024)
  const float* w2       = (const float*)d_in[12];  // (8,1024,4096)
  float* out = (float*)d_out;                      // (2048,1024) f32

  char* ws = (char*)d_ws;
  size_t off = 0;
  auto carve = [&](size_t bytes) -> char* {
    char* p = ws + off;
    off = (off + bytes + 255) & ~(size_t)255;
    return p;
  };
  float*  xn      = (float*) carve((size_t)SEQ_ * DIM_ * 4);
  bf16_t* xn_b    = (bf16_t*)carve((size_t)SEQ_ * DIM_ * 2);
  float*  xr      = (float*) carve((size_t)SEQ_ * 2 * DINNER * 4);
  float*  xc      = (float*) carve((size_t)SEQ_ * DINNER * 4);
  bf16_t* xc_b    = (bf16_t*)carve((size_t)SEQ_ * DINNER * 2);
  float*  xdbl    = (float*) carve((size_t)SEQ_ * XPOUT * 4);
  bf16_t* dt_b    = (bf16_t*)carve((size_t)SEQ_ * DTRANK * 2);
  float*  delta   = (float*) carve((size_t)SEQ_ * DINNER * 4);
  float*  ysc     = (float*) carve((size_t)SEQ_ * DINNER * 4);
  bf16_t* y_b     = (bf16_t*)carve((size_t)SEQ_ * DINNER * 2);
  float*  xm2     = (float*) carve((size_t)SEQ_ * DIM_ * 4);
  float*  xmn     = (float*) carve((size_t)SEQ_ * DIM_ * 4);
  bf16_t* xmn_b   = (bf16_t*)carve((size_t)SEQ_ * DIM_ * 2);
  float*  coef    = (float*) carve((size_t)SEQ_ * NEXP * 4);
  int*    counts  = (int*)   carve((size_t)NEXP * 4);
  int*    lists   = (int*)   carve((size_t)NEXP * SEQ_ * 4);
  float*  hbuf    = (float*) carve((size_t)SEQ_ * HID_ * 4);
  bf16_t* hbuf_b  = (bf16_t*)carve((size_t)SEQ_ * HID_ * 2);
  // bf16 weight staging
  bf16_t* in_wb    = (bf16_t*)carve((size_t)2 * DINNER * DIM_ * 2);
  bf16_t* xproj_wb = (bf16_t*)carve((size_t)XPOUT * DINNER * 2);
  bf16_t* dtproj_wb= (bf16_t*)carve((size_t)DINNER * DTRANK * 2);
  bf16_t* out_wb   = (bf16_t*)carve((size_t)DIM_ * DINNER * 2);
  bf16_t* w1b      = (bf16_t*)carve((size_t)NEXP * HID_ * DIM_ * 2);
  bf16_t* w2b      = (bf16_t*)carve((size_t)NEXP * DIM_ * HID_ * 2);

  auto cvt = [&](const float* src, bf16_t* dst, long n) {
    cvt_bf16_kernel<<<(unsigned)((n / 4 + 255) / 256), 256, 0, stream>>>(src, dst, n);
  };
  cvt(in_w,     in_wb,     (long)2 * DINNER * DIM_);
  cvt(xproj_w,  xproj_wb,  (long)XPOUT * DINNER);
  cvt(dtproj_w, dtproj_wb, (long)DINNER * DTRANK);
  cvt(out_w,    out_wb,    (long)DIM_ * DINNER);
  cvt(w1,       w1b,       (long)NEXP * HID_ * DIM_);
  cvt(w2,       w2b,       (long)NEXP * DIM_ * HID_);

  // 1) xn = rmsnorm(x)
  addnorm_kernel<<<SEQ_, 256, 0, stream>>>(x, nullptr, xn, xn_b);

  // 2) xr = xn @ in_proj_w^T  (2048 x 8192, K=1024)
  gemm_bf16_wmma<8><<<dim3(SEQ_ / 64, (2 * DINNER) / 128), 128, 0, stream>>>(
      xn_b, in_wb, xr, SEQ_, 2 * DINNER, DIM_,
      nullptr, nullptr, nullptr, nullptr, 0, 0);

  // 3) xc = silu(depthwise_conv(xm))
  conv_silu_kernel<<<(SEQ_ * DINNER) / 256, 256, 0, stream>>>(
      xr, conv_w, conv_b, xc, xc_b);

  // 4) x_dbl = xc @ x_proj_w^T  (2048 x 96, K=4096)
  gemm_bf16_wmma<2><<<dim3(SEQ_ / 64, XPOUT / 32), 128, 0, stream>>>(
      xc_b, xproj_wb, xdbl, SEQ_, XPOUT, DINNER,
      nullptr, nullptr, nullptr, nullptr, 0, 0);

  // 5-7) delta = softplus(dt @ dt_proj_w^T + b)
  cvt_dt_kernel<<<(SEQ_ * DTRANK) / 256, 256, 0, stream>>>(xdbl, dt_b);
  gemm_bf16_wmma<8><<<dim3(SEQ_ / 64, DINNER / 128), 128, 0, stream>>>(
      dt_b, dtproj_wb, delta, SEQ_, DINNER, DTRANK,
      nullptr, nullptr, nullptr, nullptr, 0, 0);
  softplus_kernel<<<(SEQ_ * DINNER) / 256, 256, 0, stream>>>(delta, dtproj_b);

  // 8) selective scan
  scan_kernel<<<DINNER / 256, 256, 0, stream>>>(xc, delta, xdbl, A_log, Dvec, ysc);

  // 9) y = y * silu(res)
  gatemul_kernel<<<(SEQ_ * DINNER) / 256, 256, 0, stream>>>(ysc, xr, y_b);

  // 10) xm2 = y @ out_proj_w^T  (2048 x 1024, K=4096)
  gemm_bf16_wmma<8><<<dim3(SEQ_ / 64, DIM_ / 128), 128, 0, stream>>>(
      y_b, out_wb, xm2, SEQ_, DIM_, DINNER,
      nullptr, nullptr, nullptr, nullptr, 0, 0);

  // 11) xmn = rmsnorm(xm2 + xn)
  addnorm_kernel<<<SEQ_, 256, 0, stream>>>(xm2, xn, xmn, xmn_b);

  // 12-14) router + skip init
  zero_counts_kernel<<<1, 32, 0, stream>>>(counts);
  router_kernel<<<SEQ_, 256, 0, stream>>>(xmn, gate_w, coef, counts, lists);
  copy_kernel<<<(SEQ_ * DIM_) / 256, 256, 0, stream>>>(x, out);

  // 15) routed top-2 experts: gather-GEMM -> GELU -> scatter-accumulate-GEMM
  for (int e = 0; e < NEXP; ++e) {
    gemm_bf16_wmma<8><<<dim3(SEQ_ / 64, HID_ / 128), 128, 0, stream>>>(
        xmn_b, w1b + (size_t)e * HID_ * DIM_, hbuf, SEQ_, HID_, DIM_,
        lists + e * SEQ_, nullptr, counts + e, nullptr, 0, 0);
    gelu_kernel<<<(SEQ_ * HID_) / 256, 256, 0, stream>>>(hbuf, hbuf_b, counts + e);
    gemm_bf16_wmma<8><<<dim3(SEQ_ / 64, DIM_ / 128), 128, 0, stream>>>(
        hbuf_b, w2b + (size_t)e * DIM_ * HID_, out, SEQ_, DIM_, HID_,
        nullptr, lists + e * SEQ_, counts + e, coef + e, NEXP, 1);
  }
}
